// EDMLoss_30099130810386
// MI455X (gfx1250) — matrix operand order repaired and checked by reading, not compiled, not run on hardware
//
#include <hip/hip_runtime.h>
#include <hip/hip_bf16.h>

// Shapes from the reference
#define B_  64
#define C_  8
#define L_  512
#define D_  256
#define T_  512
#define K_  512

#define AROW 264   // padded LDS row stride (bf16 elems): 256 + 8, keeps 16B alignment,
                   // row advance = 132 dwords -> 4-bank step -> conflict-free ds_b128

typedef __attribute__((ext_vector_type(16))) __bf16 v16bf;
typedef __attribute__((ext_vector_type(8)))  __bf16 v8bf;
typedef __attribute__((ext_vector_type(8)))  float  v8f;

// ---------------- workspace layout (bytes) ----------------
// [0   .. 15  ] float acc[4]   {rec_sum, mem_sum, dhat_sum, pad}
// [256 .. 2303] float mnorm[512]
// [4096.. +256K] __bf16 Mt[512*256]  (codeword-major: Mt[k][d])

__global__ void k_init(float* acc) {
  if (threadIdx.x < 4) acc[threadIdx.x] = 0.0f;
}

// One block per codeword k: transpose M[D,K] -> Mt[k][d] (bf16), compute ||m_k||^2.
__global__ void k_prep_M(const float* __restrict__ M, __bf16* __restrict__ Mt,
                         float* __restrict__ mnorm) {
  const int k = blockIdx.x;
  const int d = threadIdx.x;
  float v = M[(size_t)d * K_ + k];
  Mt[(size_t)k * D_ + d] = (__bf16)v;
  __shared__ float s[256];
  s[d] = v * v;
  __syncthreads();
  for (int off = 128; off > 0; off >>= 1) {
    if (d < off) s[d] += s[d + off];
    __syncthreads();
  }
  if (d == 0) mnorm[k] = s[0];
}

__global__ void k_sqdiff(const float* __restrict__ a, const float* __restrict__ b,
                         int n, float* acc) {
  __shared__ float s[256];
  float v = 0.f;
  for (int i = blockIdx.x * blockDim.x + threadIdx.x; i < n; i += gridDim.x * blockDim.x) {
    float d = a[i] - b[i];
    v += d * d;
  }
  s[threadIdx.x] = v;
  __syncthreads();
  for (int off = 128; off > 0; off >>= 1) {
    if (threadIdx.x < off) s[threadIdx.x] += s[threadIdx.x + off];
    __syncthreads();
  }
  if (threadIdx.x == 0) atomicAdd(acc, s[0]);
}

__global__ void k_sum(const float* __restrict__ a, int n, float* acc) {
  __shared__ float s[256];
  float v = 0.f;
  for (int i = blockIdx.x * blockDim.x + threadIdx.x; i < n; i += gridDim.x * blockDim.x)
    v += a[i];
  s[threadIdx.x] = v;
  __syncthreads();
  for (int off = 128; off > 0; off >>= 1) {
    if (threadIdx.x < off) s[threadIdx.x] += s[threadIdx.x + off];
    __syncthreads();
  }
  if (threadIdx.x == 0) atomicAdd(acc, s[0]);
}

// VQ kernel: block = one (b, t0) tile of 16 rows; 8 waves x 4 codeword-tiles = all 512 codewords.
// Per row: contrib = 2 * ( ||h||^2 + min_k( ||m_k||^2 - 2 h.m_k ) )
__global__ void __launch_bounds__(256)
k_vq(const float* __restrict__ H, const __bf16* __restrict__ Mt,
     const float* __restrict__ mnorm, float* mem_acc) {
  __shared__ __align__(16) __bf16 Abuf[16 * AROW];  // A tile: row m(t), col d, bf16
  __shared__ float psum[256];                       // partial ||h||^2
  __shared__ float wmins[8 * 16];                   // per-wave row minima
  __shared__ float crow[16];

  const int tid = threadIdx.x;
  const int b   = blockIdx.x >> 5;          // T_/16 = 32 tiles per batch
  const int t0  = (blockIdx.x & 31) << 4;

  // ---- stage H tile [16 t x 256 d] into LDS (f32 -> bf16), row sumsq in f32 ----
  const int m  = tid & 15;    // t within tile
  const int db = tid >> 4;    // d base 0..15
  float ss = 0.f;
  const float* Hb = H + (size_t)b * D_ * T_ + t0 + m;
  #pragma unroll
  for (int i = 0; i < 16; ++i) {
    const int d = db + (i << 4);
    const float v = Hb[(size_t)d * T_];
    ss += v * v;
    Abuf[m * AROW + d] = (__bf16)v;
  }
  psum[tid] = ss;
  __syncthreads();

  // ---- WMMA GEMM: dot(h_row, m_k) for this wave's 4 column tiles ----
  const int lane = tid & 31;
  const int wave = tid >> 5;
  const int n    = lane & 15;
  const int kb1  = (lane >> 4) << 4;   // A-fragment K sub-offset (0 or 16)
  const int kb2  = (lane >> 4) << 4;   // B-fragment K sub-offset (0 or 16)

  v8f acc[4];
  #pragma unroll
  for (int c = 0; c < 4; ++c) acc[c] = (v8f){0,0,0,0,0,0,0,0};

  #pragma unroll
  for (int ks = 0; ks < 8; ++ks) {     // 8 k-panels of 32 over D=256
    // A fragment (contiguous k order, same as B: lane m holds K = kb1..kb1+15)
    const __bf16* ap = &Abuf[(lane & 15) * AROW + (ks << 5) + kb1];
    union { v16bf v; v8bf h[2]; } au;
    au.h[0] = *(const v8bf*)ap;
    au.h[1] = *(const v8bf*)(ap + 8);
    const v16bf afrag = au.v;

    #pragma unroll
    for (int c = 0; c < 4; ++c) {
      const int k0 = (wave << 6) + (c << 4);
      // B fragment: lane (n, half) holds Mt[k0+n][ks*32+kb2 .. +15], contiguous 32B
      const __bf16* bp = Mt + (size_t)(k0 + n) * D_ + (ks << 5) + kb2;
      const v16bf bfrag = *(const v16bf*)bp;
      acc[c] = __builtin_amdgcn_wmma_f32_16x16x32_bf16(
          false, afrag, false, bfrag, (short)0, acc[c], false, false);
    }
  }

  // ---- val = ||m||^2 - 2*dot ; min over column tiles, then over N lanes ----
  float tmin[8];
  #pragma unroll
  for (int j = 0; j < 8; ++j) tmin[j] = 3.4e38f;
  #pragma unroll
  for (int c = 0; c < 4; ++c) {
    const float mn = mnorm[(wave << 6) + (c << 4) + n];
    #pragma unroll
    for (int j = 0; j < 8; ++j)
      tmin[j] = fminf(tmin[j], mn - 2.0f * acc[c][j]);
  }
  // butterfly min within each 16-lane half (C layout: lanes 0-15 -> rows j, 16-31 -> rows j+8)
  #pragma unroll
  for (int mask = 1; mask <= 8; mask <<= 1) {
    #pragma unroll
    for (int j = 0; j < 8; ++j)
      tmin[j] = fminf(tmin[j], __shfl_xor(tmin[j], mask, 32));
  }
  if ((lane & 15) == 0) {
    const int rbase = (lane >> 4) << 3;
    #pragma unroll
    for (int j = 0; j < 8; ++j) wmins[wave * 16 + rbase + j] = tmin[j];
  }
  __syncthreads();

  // ---- combine across waves; per-row contribution ----
  if (tid < 16) {
    float rs = 0.f;
    #pragma unroll
    for (int j = 0; j < 16; ++j) rs += psum[tid + (j << 4)];
    float rm = wmins[tid];
    #pragma unroll
    for (int w = 1; w < 8; ++w) rm = fminf(rm, wmins[w * 16 + tid]);
    crow[tid] = 2.0f * (rs + rm);
  }
  __syncthreads();
  if (tid == 0) {
    float tot = 0.f;
    #pragma unroll
    for (int i = 0; i < 16; ++i) tot += crow[i];
    atomicAdd(mem_acc, tot);
  }
}

__global__ void k_final(const float* __restrict__ acc, float* __restrict__ out) {
  // rec/(B*C*L) + 0.25*mem/(B*D*T) - 0.1*dhat/(B*T)
  out[0] = acc[0] * (1.0f / 262144.0f)
         + 0.25f * acc[1] * (1.0f / 8388608.0f)
         - 0.1f  * acc[2] * (1.0f / 32768.0f);
}

extern "C" void kernel_launch(void* const* d_in, const int* in_sizes, int n_in,
                              void* d_out, int out_size, void* d_ws, size_t ws_size,
                              hipStream_t stream) {
  const float* Xhat = (const float*)d_in[0];
  const float* X    = (const float*)d_in[1];
  const float* H    = (const float*)d_in[2];
  const float* M    = (const float*)d_in[3];
  const float* Dhat = (const float*)d_in[4];
  (void)in_sizes; (void)n_in; (void)out_size; (void)ws_size;

  char*   ws    = (char*)d_ws;
  float*  acc   = (float*)ws;             // 4 floats
  float*  mnorm = (float*)(ws + 256);     // 512 floats
  __bf16* Mt    = (__bf16*)(ws + 4096);   // 512*256 bf16 = 256 KB (L2-resident)

  k_init  <<<1,   64,  0, stream>>>(acc);
  k_prep_M<<<K_,  256, 0, stream>>>(M, Mt, mnorm);
  k_sqdiff<<<256, 256, 0, stream>>>(Xhat, X, B_ * C_ * L_, acc + 0);
  k_sum   <<<32,  256, 0, stream>>>(Dhat, B_ * T_, acc + 2);
  k_vq    <<<B_ * (T_ / 16), 256, 0, stream>>>(H, Mt, mnorm, acc + 1);
  k_final <<<1,   1,   0, stream>>>(acc, (float*)d_out);
}
